// Quantize_37512244363882
// MI455X (gfx1250) — compile-verified, matrix operand order checked by reading, not run on hardware
//
#include <hip/hip_runtime.h>
#include <hip/hip_bf16.h>
#include <stdint.h>

typedef __attribute__((ext_vector_type(16))) __bf16        v16bf;
typedef __attribute__((ext_vector_type(8)))  float         v8f;
typedef __attribute__((ext_vector_type(8)))  int           v8i;
typedef __attribute__((ext_vector_type(4)))  int           v4i;
typedef __attribute__((ext_vector_type(4)))  unsigned int  v4u;

#define Bc 8
#define Tc 32768
#define Dc 64
#define Kc 512

__device__ __forceinline__ unsigned short f32_to_bf16(float f) {
  unsigned int u = __builtin_bit_cast(unsigned int, f);
  unsigned int r = u + 0x7FFFu + ((u >> 16) & 1u);   // round-to-nearest-even
  return (unsigned short)(r >> 16);
}
__device__ __forceinline__ float bf16_to_f32(unsigned short s) {
  unsigned int u = ((unsigned int)s) << 16;
  return __builtin_bit_cast(float, u);
}

// ---------------------------------------------------------------------------
// Preprocess: normalized codebooks -> bf16 B-fragments in WMMA lane layout,
// per-code -0.5*|en|^2 (used directly as WMMA C-init), zero diff accumulator.
// B-fragment layout (16-bit B 32x16, mirrors ISA A 16x32 table):
//   lane l: n = l&15, group = l>>4
//   half j: v=j>>1, jj=j&1, kk = (v>>2)*16 + (v&3)*2 + jj + group*8
// Fragment id = (table*32 + ktile)*2 + dchunk; element e = (frag*32+lane)*16+j
// ---------------------------------------------------------------------------
__global__ void vq_preprocess(const float* __restrict__ W, const int* __restrict__ sel,
                              float* __restrict__ cn05, unsigned short* __restrict__ bfrag,
                              float* __restrict__ diff_acc) {
  const int tid = blockIdx.x * blockDim.x + threadIdx.x;
  float c0 = 0.f, c1 = 0.f;
  for (int i = 0; i < Bc; ++i) { if (sel[i] == 0) c0 += 1.f; else c1 += 1.f; }
  if (tid == 0) *diff_acc = 0.f;

  const int total = 2 * 32 * 2 * 32 * 16;   // 65536 bf16 fragment elements
  for (int e = tid; e < total; e += gridDim.x * blockDim.x) {
    int j    = e & 15;
    int lane = (e >> 4) & 31;
    int rest = e >> 9;                      // fragment id
    int dc   = rest & 1;
    int kt   = (rest >> 1) & 31;
    int w    = rest >> 6;
    int n = lane & 15, group = lane >> 4;
    int v = j >> 1, jj = j & 1;
    int kk = ((v >> 2) << 4) + ((v & 3) << 1) + jj + (group << 3);
    int d = dc * 32 + kk;
    int code = kt * 16 + n;
    float w0 = W[(0 * Kc + code) * Dc + d];
    float w1 = W[(1 * Kc + code) * Dc + d];
    float rn = rsqrtf(c0 * w0 * w0 + c1 * w1 * w1);   // batch-axis norm
    float val = (w == 0 ? w0 : w1) * rn;
    bfrag[e] = f32_to_bf16(val);
  }
  if (tid < 2 * Kc) {
    int w = tid >> 9;
    int code = tid & (Kc - 1);
    float s = 0.f;
    for (int d = 0; d < Dc; ++d) {
      float w0 = W[(0 * Kc + code) * Dc + d];
      float w1 = W[(1 * Kc + code) * Dc + d];
      float rn = rsqrtf(c0 * w0 * w0 + c1 * w1 * w1);
      float val = bf16_to_f32(f32_to_bf16((w == 0 ? w0 : w1) * rn));
      s += val * val;
    }
    cn05[tid] = -0.5f * s;   // C-init: score m = x.en - 0.5|en|^2 ; argmax(m)
  }
}

// ---------------------------------------------------------------------------
// Main: block = 256 rows of one batch, 8 waves x 2 M-tiles.
// B table (64KB) staged to LDS once per block via TDM tensor_load_to_lds.
// k-loop unrolled x2 with SSA ping-pong double buffering (no rotation copies).
// ---------------------------------------------------------------------------
__global__ __launch_bounds__(256) void vq_main(
    const float* __restrict__ x, const float* __restrict__ W,
    const int* __restrict__ sel, const float* __restrict__ cn05,
    const unsigned short* __restrict__ bfrag,
    float* __restrict__ out, float* __restrict__ diff_acc) {
  extern __shared__ __align__(16) char smem[];
  float*          xs  = (float*)smem;                        // 64 KB: x tile f32
  unsigned short* Bs  = (unsigned short*)(smem + 65536);     // 64 KB: B fragments
  int*            bks = (int*)(smem + 131072);               // 1 KB : winners
  float*          cns = (float*)(smem + 132096);             // 2 KB : -0.5|en|^2

  const int b     = blockIdx.y;
  const int t0    = blockIdx.x * 256;
  const int tid   = threadIdx.x;
  const int table = __builtin_amdgcn_readfirstlane(sel[b]);

  // --- TDM: one descriptor moves this table's 64KB fragment buffer to LDS ---
  if ((tid >> 5) == 0) {
    size_t ga = (size_t)bfrag + (size_t)table * 65536;
    v4u g0; v8i g1; v4i g2 = {0, 0, 0, 0}, g3 = {0, 0, 0, 0};
    v8i g4 = {0, 0, 0, 0, 0, 0, 0, 0};
    g0[0] = 1u;                                        // count=1 (valid D#)
    g0[1] = 65536u;                                    // lds_addr (bytes)
    g0[2] = (unsigned)(ga & 0xFFFFFFFFu);              // global_addr[31:0]
    g0[3] = (unsigned)((ga >> 32) & 0x01FFFFFFu)       // global_addr[56:32]
          | (2u << 30);                                // type=2 ("image")
    g1[0] = (int)(2u << 16);                           // data_size=4B
    g1[1] = (int)(16384u << 16);                       // tensor_dim0 = 16384 dwords
    g1[2] = (int)(1u << 16);                           // tensor_dim1 = 1
    g1[3] = (int)(16384u << 16);                       // tile_dim0 = 16384
    g1[4] = 1;                                         // tile_dim1 = 1
    g1[5] = 16384;                                     // tensor_dim0_stride
    g1[6] = 0; g1[7] = 0;
    __builtin_amdgcn_tensor_load_to_lds(g0, g1, g2, g3, g4, 0);
  }

  // --- stage x tile (single HBM read of x) and cn table ---
  const float4* xin = (const float4*)(x + ((size_t)b * Tc + t0) * Dc);
  float4* xsv = (float4*)xs;
  for (int i = tid; i < 256 * Dc / 4; i += 256) xsv[i] = xin[i];
  for (int i = tid; i < Kc; i += 256) cns[i] = cn05[table * Kc + i];

  if ((tid >> 5) == 0) __builtin_amdgcn_s_wait_tensorcnt(0);
  __syncthreads();

  const int lane  = tid & 31;
  const int group = lane >> 4;
  const int nl    = lane & 15;
  const int wave  = tid >> 5;
  const int r0    = wave * 32;

  // --- build A fragments for two 16-row M-tiles (one-time, from LDS) ---
  v16bf A0[2], A1[2];
#pragma unroll
  for (int dcn = 0; dcn < 2; ++dcn) {
    v8i pa, pb;
#pragma unroll
    for (int v = 0; v < 8; ++v) {
      int kk = ((v >> 2) << 4) + ((v & 3) << 1) + (group << 3);
      int d  = dcn * 32 + kk;
      const float* p0 = &xs[(r0 + nl) * Dc + d];
      const float* p1 = &xs[(r0 + 16 + nl) * Dc + d];
      pa[v] = (unsigned)f32_to_bf16(p0[0]) | ((unsigned)f32_to_bf16(p0[1]) << 16);
      pb[v] = (unsigned)f32_to_bf16(p1[0]) | ((unsigned)f32_to_bf16(p1[1]) << 16);
    }
    A0[dcn] = __builtin_bit_cast(v16bf, pa);
    A1[dcn] = __builtin_bit_cast(v16bf, pb);
  }

  float best0[8], best1[8];
  int   bk0[8],   bk1[8];
#pragma unroll
  for (int v = 0; v < 8; ++v) {
    best0[v] = -3.4e38f; best1[v] = -3.4e38f; bk0[v] = 0; bk1[v] = 0;
  }

  const v8i* Bv = (const v8i*)Bs + lane;   // per-lane 32B within each 1KB frag

  auto step = [&](int kt, v8i rb0, v8i rb1, float ci) {
    v16bf B0 = __builtin_bit_cast(v16bf, rb0);
    v16bf B1 = __builtin_bit_cast(v16bf, rb1);
    v8f c0 = {ci, ci, ci, ci, ci, ci, ci, ci};
    v8f c1 = c0;
    c0 = __builtin_amdgcn_wmma_f32_16x16x32_bf16(false, A0[0], false, B0, (short)0, c0, false, false);
    c1 = __builtin_amdgcn_wmma_f32_16x16x32_bf16(false, A1[0], false, B0, (short)0, c1, false, false);
    c0 = __builtin_amdgcn_wmma_f32_16x16x32_bf16(false, A0[1], false, B1, (short)0, c0, false, false);
    c1 = __builtin_amdgcn_wmma_f32_16x16x32_bf16(false, A1[1], false, B1, (short)0, c1, false, false);
    const int kidx = kt * 16 + nl;
#pragma unroll
    for (int v = 0; v < 8; ++v) {
      if (c0[v] > best0[v]) { best0[v] = c0[v]; bk0[v] = kidx; }  // strict > == first-min of dist
      if (c1[v] > best1[v]) { best1[v] = c1[v]; bk1[v] = kidx; }
    }
  };

  // SSA ping-pong: each half computes with its set, then reloads that (dead)
  // set for kt+2 — loop-carried values need no register copies.
  v8i   a0 = Bv[0],  a1 = Bv[32];  float ca = cns[nl];
  v8i   b0 = Bv[64], b1 = Bv[96];  float cb = cns[16 + nl];
#pragma unroll 1
  for (int kt = 0; kt < 32; kt += 2) {
    step(kt, a0, a1, ca);
    {
      const int kn = (kt + 2) & 31;           // tail wraps harmlessly
      a0 = Bv[kn * 64]; a1 = Bv[kn * 64 + 32]; ca = cns[kn * 16 + nl];
    }
    __builtin_amdgcn_sched_barrier(0);        // keep prefetch ahead of next half
    step(kt + 1, b0, b1, cb);
    {
      const int kn = (kt + 3) & 31;
      b0 = Bv[kn * 64]; b1 = Bv[kn * 64 + 32]; cb = cns[kn * 16 + nl];
    }
    __builtin_amdgcn_sched_barrier(0);
  }

  // --- argmax across the 16 lanes holding the same M row ---
#pragma unroll
  for (int off = 1; off < 16; off <<= 1) {
#pragma unroll
    for (int v = 0; v < 8; ++v) {
      float os0 = __shfl_xor(best0[v], off, 32);
      int   ok0 = __shfl_xor(bk0[v],   off, 32);
      if (os0 > best0[v] || (os0 == best0[v] && ok0 < bk0[v])) { best0[v] = os0; bk0[v] = ok0; }
      float os1 = __shfl_xor(best1[v], off, 32);
      int   ok1 = __shfl_xor(bk1[v],   off, 32);
      if (os1 > best1[v] || (os1 == best1[v] && ok1 < bk1[v])) { best1[v] = os1; bk1[v] = ok1; }
    }
  }
  if (nl == 0) {                    // C layout: lanes>=16 hold M=v+8
#pragma unroll
    for (int v = 0; v < 8; ++v) {
      bks[r0 + group * 8 + v]      = bk0[v];
      bks[r0 + 16 + group * 8 + v] = bk1[v];
    }
  }
  __syncthreads();

  // --- gather raw codes, write out (== quantize), accumulate diff ---
  const int row = tid;              // 256 rows, 1 thread each
  const int k   = bks[row];
  const float4* wq = (const float4*)(W + ((size_t)table * Kc + k) * Dc);
  const float4* xg = (const float4*)(xs + row * Dc);
  float4* og = (float4*)(out + ((size_t)b * Tc + t0 + row) * Dc);
  float acc = 0.f;
#pragma unroll
  for (int i = 0; i < 16; ++i) {
    float4 q  = wq[i];
    float4 xv = xg[i];
    og[i] = q;
    float dx = q.x - xv.x, dy = q.y - xv.y, dz = q.z - xv.z, dw = q.w - xv.w;
    acc += dx * dx + dy * dy + dz * dz + dw * dw;
  }
#pragma unroll
  for (int off = 16; off >= 1; off >>= 1) acc += __shfl_xor(acc, off, 32);
  if (lane == 0) atomicAdd(diff_acc, acc);
}

__global__ void vq_finalize(const float* __restrict__ diff_acc,
                            const int* __restrict__ sel,
                            float* __restrict__ out_tail) {
  int t = threadIdx.x;
  if (t == 0) out_tail[0] = *diff_acc * (1.0f / ((float)Bc * (float)Tc * (float)Dc));
  if (t < Bc) out_tail[1 + t] = (float)sel[t];
}

extern "C" void kernel_launch(void* const* d_in, const int* in_sizes, int n_in,
                              void* d_out, int out_size, void* d_ws, size_t ws_size,
                              hipStream_t stream) {
  const float* x  = (const float*)d_in[0];   // [8,32768,64] f32
  const float* W  = (const float*)d_in[1];   // [2,512,64]   f32
  const int* sel  = (const int*)d_in[2];     // [8]          i32
  float* out      = (float*)d_out;

  float* diff_acc       = (float*)d_ws;                            // 4 B
  float* cn05           = (float*)((char*)d_ws + 256);             // 4 KB
  unsigned short* bfrag = (unsigned short*)((char*)d_ws + 8192);   // 128 KB

  vq_preprocess<<<64, 256, 0, stream>>>(W, sel, cn05, bfrag, diff_acc);
  vq_main<<<dim3(Tc / 256, Bc), 256, 134144, stream>>>(x, W, sel, cn05, bfrag,
                                                       out, diff_acc);
  vq_finalize<<<1, 32, 0, stream>>>(diff_acc, sel, out + (size_t)Bc * Tc * Dc);
}
